// PersistentWorldMemoryBank_29953101922438
// MI455X (gfx1250) — matrix operand order repaired and checked by reading, not compiled, not run on hardware
//
#include <hip/hip_runtime.h>
#include <hip/hip_bf16.h>

#define N_OBJ   4096
#define D_MEM   256
#define D_ROI   1024
#define D_STRUCT 256
#define D_CAM   128
#define D_MOT   64
#define GRU_IN  704
#define GRU_OUT 768
#define KV_PAD  4224   // 4097 KV rows padded to a multiple of 128 (and 64)
#define KV_VALID 4097
#define NHEAD   4

typedef __bf16 bf16x16 __attribute__((ext_vector_type(16)));
typedef float  f32x8   __attribute__((ext_vector_type(8)));

union FragBF { uint4 u[2]; bf16x16 v; };

__device__ __forceinline__ unsigned short f2bf(float f) {
  union { float f; unsigned int u; } c; c.f = f;
  unsigned int u = c.u;
  return (unsigned short)((u + 0x7FFFu + ((u >> 16) & 1u)) >> 16);  // RNE
}

__device__ __forceinline__ f32x8 zero8() {
  f32x8 z = {0.f, 0.f, 0.f, 0.f, 0.f, 0.f, 0.f, 0.f};
  return z;
}

__device__ __forceinline__ f32x8 wmma_bf16(const FragBF& a, const FragBF& b, f32x8 c) {
#if defined(__AMDGCN__)
  return __builtin_amdgcn_wmma_f32_16x16x32_bf16(false, a.v, false, b.v, (short)0, c,
                                                 false, false);
#else
  return c;
#endif
}

enum { EPI_BIAS = 0, EPI_RELU = 1, EPI_MIX = 2 };

// ---------------------------------------------------------------------------
// Generic bf16-WMMA GEMM: D[M x N] = epilogue(A[M x K] * B[K x N] + bias)
// Block tile 128x128, 8 waves (4 along M x 2 along N), k-step 32.
// Double-buffered LDS staging (one barrier per k-step) + global prefetch of
// the tile after next.  M mult of 128, N mult of 128, K mult of 32.
// ---------------------------------------------------------------------------
__global__ __launch_bounds__(256) void gemm_bf16_k(
    const float* __restrict__ A, int lda,
    const float* __restrict__ B, int ldb,
    const float* __restrict__ bias,
    float* __restrict__ D, int ldd,
    int K, int mode,
    const int* __restrict__ vm, const float* __restrict__ visbuf)
{
  __shared__ unsigned short As[2][128][32] __attribute__((aligned(16)));
  __shared__ unsigned short Bs[2][128][32] __attribute__((aligned(16)));  // Bs[.][n][k]
  const int tid = threadIdx.x;
  const int wave = tid >> 5, lane = tid & 31;
  const int l15 = lane & 15, half = lane >> 4;
  const int bm = blockIdx.x * 128, bn = blockIdx.y * 128;
  const int wm = (wave >> 1) * 32, wn = (wave & 1) * 64;

  const int arow = tid >> 1, acol = (tid & 1) * 16;
  const int bkr  = tid >> 3, bn0  = (tid & 7) * 16;
  const float* aptr = A + (size_t)(bm + arow) * lda + acol;
  const float* bptr = B + (size_t)bkr * ldb + bn + bn0;

  auto stage = [&](int buf, int k0) {
    const float* asrc = aptr + k0;
#pragma unroll
    for (int i = 0; i < 16; i += 4) {
      float4 f = *(const float4*)(asrc + i);
      As[buf][arow][acol + i + 0] = f2bf(f.x);
      As[buf][arow][acol + i + 1] = f2bf(f.y);
      As[buf][arow][acol + i + 2] = f2bf(f.z);
      As[buf][arow][acol + i + 3] = f2bf(f.w);
    }
    const float* bsrc = bptr + (size_t)k0 * ldb;
#pragma unroll
    for (int i = 0; i < 16; i += 4) {
      float4 f = *(const float4*)(bsrc + i);
      Bs[buf][bn0 + i + 0][bkr] = f2bf(f.x);
      Bs[buf][bn0 + i + 1][bkr] = f2bf(f.y);
      Bs[buf][bn0 + i + 2][bkr] = f2bf(f.z);
      Bs[buf][bn0 + i + 3][bkr] = f2bf(f.w);
    }
  };

  f32x8 acc[2][4];
#pragma unroll
  for (int mi = 0; mi < 2; ++mi)
#pragma unroll
    for (int ni = 0; ni < 4; ++ni) acc[mi][ni] = zero8();

  const int nk = K >> 5;
  stage(0, 0);
  __syncthreads();

  for (int t = 0; t < nk; ++t) {
    if (t + 2 < nk) {  // prefetch tile after next toward the WGP
      __builtin_prefetch(aptr + (t + 2) * 32, 0, 3);
      __builtin_prefetch(bptr + (size_t)((t + 2) * 32) * ldb, 0, 3);
    }
    if (t + 1 < nk) stage((t + 1) & 1, (t + 1) * 32);  // fill other buffer

    const int buf = t & 1;
    FragBF afrag[2], bfrag[4];
#pragma unroll
    for (int mi = 0; mi < 2; ++mi) {
      int r = wm + mi * 16 + l15;
      afrag[mi].u[0] = *(const uint4*)&As[buf][r][half * 8];       // K 0..7 / 8..15
      afrag[mi].u[1] = *(const uint4*)&As[buf][r][16 + half * 8];  // K 16..23 / 24..31
    }
#pragma unroll
    for (int ni = 0; ni < 4; ++ni) {
      int r = wn + ni * 16 + l15;
      bfrag[ni].u[0] = *(const uint4*)&Bs[buf][r][half * 8];       // K 0..7 / 8..15
      bfrag[ni].u[1] = *(const uint4*)&Bs[buf][r][16 + half * 8];  // K 16..23 / 24..31
    }
#pragma unroll
    for (int mi = 0; mi < 2; ++mi)
#pragma unroll
      for (int ni = 0; ni < 4; ++ni)
        acc[mi][ni] = wmma_bf16(afrag[mi], bfrag[ni], acc[mi][ni]);
    __syncthreads();  // staging of buf^1 visible; reads of buf done
  }

#pragma unroll
  for (int mi = 0; mi < 2; ++mi)
#pragma unroll
    for (int ni = 0; ni < 4; ++ni) {
      int col = bn + wn + ni * 16 + l15;
      float bsv = bias[col];
#pragma unroll
      for (int r = 0; r < 8; ++r) {
        int row = bm + wm + mi * 16 + r + half * 8;
        float v = acc[mi][ni][r] + bsv;
        if (mode == EPI_RELU) {
          v = fmaxf(v, 0.f);
        } else if (mode == EPI_MIX) {
          if (vm[row] != 0) v = visbuf[(size_t)row * 256 + col];  // seen branch
        }
        D[(size_t)row * ldd + col] = v;
      }
    }
}

// ---------------------------------------------------------------------------
// Row-wise LayerNorm over 256 columns, in place. One block per row.
// ---------------------------------------------------------------------------
__global__ __launch_bounds__(256) void layernorm_k(float* __restrict__ x,
    const float* __restrict__ g, const float* __restrict__ b)
{
  __shared__ float red[256];
  int row = blockIdx.x, t = threadIdx.x;
  float v = x[(size_t)row * 256 + t];
  red[t] = v; __syncthreads();
  for (int s = 128; s > 0; s >>= 1) { if (t < s) red[t] += red[t + s]; __syncthreads(); }
  float mu = red[0] * (1.f / 256.f);
  __syncthreads();
  float d = v - mu;
  red[t] = d * d; __syncthreads();
  for (int s = 128; s > 0; s >>= 1) { if (t < s) red[t] += red[t + s]; __syncthreads(); }
  float var = red[0] * (1.f / 256.f);
  x[(size_t)row * 256 + t] = d * rsqrtf(var + 1e-5f) * g[t] + b[t];
}

// ---------------------------------------------------------------------------
// Fill KV rows 4096..4223: row 4096 = ego projection, rest = 0.
// ---------------------------------------------------------------------------
__global__ __launch_bounds__(256) void ego_pad_k(const float* __restrict__ ego_tok,
    const float* __restrict__ ego_w, const float* __restrict__ ego_b,
    float* __restrict__ kv)
{
  int b = blockIdx.x, t = threadIdx.x;
  if (b == 0) {
    float s = ego_b[t];
    for (int k = 0; k < D_CAM; ++k) s += ego_tok[k] * ego_w[k * 256 + t];
    kv[(size_t)N_OBJ * 256 + t] = s;
  } else {
    kv[(size_t)(N_OBJ + b) * 256 + t] = 0.f;
  }
}

// ---------------------------------------------------------------------------
// Attention pass 1: per (head, q-row) softmax stats (max, sum-exp) via WMMA QK^T.
// Block: 64 q rows of one head; 8 waves = 4 row-strips x 2 key-halves of 128 keys.
// Double-buffered key staging, hoisted Q fragments, prefetch.
// ---------------------------------------------------------------------------
__global__ __launch_bounds__(256) void attn_stats_k(
    const float* __restrict__ Q, const float* __restrict__ Km, float* __restrict__ stats)
{
  __shared__ unsigned short qa[64][64] __attribute__((aligned(16)));
  __shared__ unsigned short kb[2][128][64] __attribute__((aligned(16)));
  __shared__ float sm[2][64], ss[2][64];
  const int h = blockIdx.y;
  const int n0 = blockIdx.x * 64;
  const int tid = threadIdx.x, wave = tid >> 5, lane = tid & 31;
  const int l15 = lane & 15, half = lane >> 4;

  { int row = tid >> 2, d0 = (tid & 3) * 16;
    const float* src = Q + (size_t)(n0 + row) * 256 + h * 64 + d0;
#pragma unroll
    for (int i = 0; i < 16; ++i) qa[row][d0 + i] = f2bf(src[i]);
  }
  const int skr = tid >> 1, sd0 = (tid & 1) * 32;
  const float* kbase = Km + (size_t)skr * 256 + h * 64 + sd0;
  auto stage_kb = [&](int buf, int kt) {
    const float* src = kbase + (size_t)kt * 128 * 256;
#pragma unroll
    for (int i = 0; i < 32; ++i) kb[buf][skr][sd0 + i] = f2bf(src[i]);
  };

  const int rowstrip = wave & 3, keyhalf = wave >> 2;
  float rm[8], rs[8];
#pragma unroll
  for (int r = 0; r < 8; ++r) { rm[r] = -1e30f; rs[r] = 0.f; }

  stage_kb(0, 0);
  __syncthreads();

  FragBF aq[2];  // Q fragments are loop-invariant
#pragma unroll
  for (int ks = 0; ks < 2; ++ks) {
    int r = rowstrip * 16 + l15;
    aq[ks].u[0] = *(const uint4*)&qa[r][ks * 32 + half * 8];
    aq[ks].u[1] = *(const uint4*)&qa[r][ks * 32 + 16 + half * 8];
  }

  const int NT = KV_PAD / 128;
  for (int kt = 0; kt < NT; ++kt) {
    if (kt + 2 < NT) __builtin_prefetch(kbase + (size_t)(kt + 2) * 128 * 256, 0, 3);
    if (kt + 1 < NT) stage_kb((kt + 1) & 1, kt + 1);
    const int buf = kt & 1;
#pragma unroll
    for (int ktile = 0; ktile < 4; ++ktile) {
      int keybase = keyhalf * 64 + ktile * 16;
      int brow = keybase + l15;
      FragBF bkf[2];
#pragma unroll
      for (int ks = 0; ks < 2; ++ks) {
        bkf[ks].u[0] = *(const uint4*)&kb[buf][brow][ks * 32 + half * 8];
        bkf[ks].u[1] = *(const uint4*)&kb[buf][brow][ks * 32 + 16 + half * 8];
      }
      f32x8 s = zero8();
      s = wmma_bf16(aq[0], bkf[0], s);
      s = wmma_bf16(aq[1], bkf[1], s);
      int keyglob = kt * 128 + keybase + l15;
#pragma unroll
      for (int r = 0; r < 8; ++r) {
        float x = (keyglob < KV_VALID) ? s[r] * 0.125f : -1e30f;   // 1/sqrt(64)
        float tm = x;
        tm = fmaxf(tm, __shfl_xor(tm, 1));
        tm = fmaxf(tm, __shfl_xor(tm, 2));
        tm = fmaxf(tm, __shfl_xor(tm, 4));
        tm = fmaxf(tm, __shfl_xor(tm, 8));
        float mnew = fmaxf(rm[r], tm);
        float p = __expf(x - mnew);
        p += __shfl_xor(p, 1);
        p += __shfl_xor(p, 2);
        p += __shfl_xor(p, 4);
        p += __shfl_xor(p, 8);
        rs[r] = rs[r] * __expf(rm[r] - mnew) + p;
        rm[r] = mnew;
      }
    }
    __syncthreads();  // kb[buf^1] staged; kb[buf] reads done
  }
  if (l15 == 0) {
#pragma unroll
    for (int r = 0; r < 8; ++r) {
      int row = rowstrip * 16 + r + half * 8;
      sm[keyhalf][row] = rm[r];
      ss[keyhalf][row] = rs[r];
    }
  }
  __syncthreads();
  if (tid < 64) {
    float m0 = sm[0][tid], m1 = sm[1][tid];
    float M = fmaxf(m0, m1);
    float S = ss[0][tid] * __expf(m0 - M) + ss[1][tid] * __expf(m1 - M);
    stats[(size_t)(h * N_OBJ + n0 + tid) * 2 + 0] = M;
    stats[(size_t)(h * N_OBJ + n0 + tid) * 2 + 1] = S;
  }
}

// ---------------------------------------------------------------------------
// Attention pass 2: recompute S, P = exp(S - max), O = P*V via WMMA; O /= denom.
// Block: 64 q rows of one head; waves = 4 row-strips x 2 dim-halves.
// ---------------------------------------------------------------------------
__global__ __launch_bounds__(256) void attn_av_k(
    const float* __restrict__ Q, const float* __restrict__ Km,
    const float* __restrict__ Vm, const float* __restrict__ stats,
    float* __restrict__ ctx)
{
  __shared__ unsigned short qa[64][64] __attribute__((aligned(16)));
  __shared__ unsigned short kb[64][64] __attribute__((aligned(16)));
  __shared__ unsigned short pt[64][64] __attribute__((aligned(16)));  // P tile (bf16)
  __shared__ unsigned short vt[64][64] __attribute__((aligned(16)));  // V^T: vt[d][key]
  __shared__ float rowm[64], rowd[64];
  const int h = blockIdx.y, n0 = blockIdx.x * 64;
  const int tid = threadIdx.x, wave = tid >> 5, lane = tid & 31;
  const int l15 = lane & 15, half = lane >> 4;

  { int row = tid >> 2, d0 = (tid & 3) * 16;
    const float* src = Q + (size_t)(n0 + row) * 256 + h * 64 + d0;
#pragma unroll
    for (int i = 0; i < 16; ++i) qa[row][d0 + i] = f2bf(src[i]);
  }
  if (tid < 64) {
    rowm[tid] = stats[(size_t)(h * N_OBJ + n0 + tid) * 2 + 0];
    rowd[tid] = stats[(size_t)(h * N_OBJ + n0 + tid) * 2 + 1];
  }
  const int rowstrip = wave & 3, dhalf = wave >> 2;
  f32x8 o[2];
  o[0] = zero8(); o[1] = zero8();

  __syncthreads();
  FragBF aq[2];  // Q fragments are loop-invariant
#pragma unroll
  for (int ks = 0; ks < 2; ++ks) {
    int r = rowstrip * 16 + l15;
    aq[ks].u[0] = *(const uint4*)&qa[r][ks * 32 + half * 8];
    aq[ks].u[1] = *(const uint4*)&qa[r][ks * 32 + 16 + half * 8];
  }

  const int skr = tid >> 2, sd0 = (tid & 3) * 16;
  const float* kbase = Km + (size_t)skr * 256 + h * 64 + sd0;
  const float* vbase = Vm + (size_t)skr * 256 + h * 64 + sd0;

  for (int kt = 0; kt < KV_PAD / 64; ++kt) {
    __syncthreads();  // prior P*V reads of kb/vt/pt done
    { const float* ksrc = kbase + (size_t)kt * 64 * 256;
      const float* vsrc = vbase + (size_t)kt * 64 * 256;
      if (kt + 1 < KV_PAD / 64) {
        __builtin_prefetch(ksrc + 64 * 256, 0, 3);
        __builtin_prefetch(vsrc + 64 * 256, 0, 3);
      }
#pragma unroll
      for (int i = 0; i < 16; ++i) kb[skr][sd0 + i] = f2bf(ksrc[i]);
#pragma unroll
      for (int i = 0; i < 16; ++i) vt[sd0 + i][skr] = f2bf(vsrc[i]);
    }
    __syncthreads();
    // S tiles for this wave's (rowstrip, key range dhalf*32..dhalf*32+31) -> P in LDS
#pragma unroll
    for (int j = 0; j < 2; ++j) {
      int keybase = dhalf * 32 + j * 16;
      int brow = keybase + l15;
      FragBF bkf[2];
#pragma unroll
      for (int ks = 0; ks < 2; ++ks) {
        bkf[ks].u[0] = *(const uint4*)&kb[brow][ks * 32 + half * 8];
        bkf[ks].u[1] = *(const uint4*)&kb[brow][ks * 32 + 16 + half * 8];
      }
      f32x8 s = zero8();
      s = wmma_bf16(aq[0], bkf[0], s);
      s = wmma_bf16(aq[1], bkf[1], s);
      int keyglob = kt * 64 + keybase + l15;
      int pcol = keybase + l15;
#pragma unroll
      for (int r = 0; r < 8; ++r) {
        int prow = rowstrip * 16 + r + half * 8;
        float p = (keyglob < KV_VALID) ? __expf(s[r] * 0.125f - rowm[prow]) : 0.f;
        pt[prow][pcol] = f2bf(p);
      }
    }
    __syncthreads();
    // O += P * V  (K-dim = 64 keys, two k-steps)
    FragBF ap[2];
#pragma unroll
    for (int ks = 0; ks < 2; ++ks) {
      int r = rowstrip * 16 + l15;
      ap[ks].u[0] = *(const uint4*)&pt[r][ks * 32 + half * 8];
      ap[ks].u[1] = *(const uint4*)&pt[r][ks * 32 + 16 + half * 8];
    }
#pragma unroll
    for (int dt = 0; dt < 2; ++dt) {
      int vrow = dhalf * 32 + dt * 16 + l15;
      FragBF bvf[2];
#pragma unroll
      for (int ks = 0; ks < 2; ++ks) {
        bvf[ks].u[0] = *(const uint4*)&vt[vrow][ks * 32 + half * 8];
        bvf[ks].u[1] = *(const uint4*)&vt[vrow][ks * 32 + 16 + half * 8];
      }
      o[dt] = wmma_bf16(ap[0], bvf[0], o[dt]);
      o[dt] = wmma_bf16(ap[1], bvf[1], o[dt]);
    }
  }
#pragma unroll
  for (int dt = 0; dt < 2; ++dt) {
    int col = dhalf * 32 + dt * 16 + l15;
#pragma unroll
    for (int r = 0; r < 8; ++r) {
      int row = rowstrip * 16 + r + half * 8;
      ctx[(size_t)(n0 + row) * 256 + h * 64 + col] = o[dt][r] * (1.f / rowd[row]);
    }
  }
}

// ---------------------------------------------------------------------------
// Pack struct/cam/mot into gru_input columns 256..703 (cols 0..255 come from MIX).
// ---------------------------------------------------------------------------
__global__ __launch_bounds__(256) void pack_k(const float* __restrict__ st,
    const float* __restrict__ cam, const float* __restrict__ mot,
    float* __restrict__ gru_in)
{
  int idx = blockIdx.x * 256 + threadIdx.x;
  if (idx >= N_OBJ * 448) return;
  int n = idx / 448, c = idx % 448;
  float v; int col;
  if (c < 256)      { v = st[(size_t)n * 256 + c];          col = 256 + c; }
  else if (c < 384) { v = cam[(size_t)n * 128 + (c - 256)]; col = 512 + (c - 256); }
  else              { v = mot[(size_t)n * 64 + (c - 384)];  col = 640 + (c - 384); }
  gru_in[(size_t)n * GRU_IN + col] = v;
}

// ---------------------------------------------------------------------------
// GRU gate combine + valid mask.  gi/gh layout: [r | z | n] blocks of 256.
// ---------------------------------------------------------------------------
__global__ __launch_bounds__(256) void gru_final_k(const float* __restrict__ gi,
    const float* __restrict__ gh, const float* __restrict__ memory,
    const int* __restrict__ valid, float* __restrict__ out)
{
  int idx = blockIdx.x * 256 + threadIdx.x;
  int n = idx >> 8, c = idx & 255;
  size_t b = (size_t)n * GRU_OUT;
  float r = 1.f / (1.f + __expf(-(gi[b + c] + gh[b + c])));
  float z = 1.f / (1.f + __expf(-(gi[b + 256 + c] + gh[b + 256 + c])));
  float nn = tanhf(gi[b + 512 + c] + r * gh[b + 512 + c]);
  float m = memory[idx];
  float hval = (1.f - z) * nn + z * m;
  out[idx] = (valid[n] != 0) ? hval : 0.f;
}

// ---------------------------------------------------------------------------
extern "C" void kernel_launch(void* const* d_in, const int* in_sizes, int n_in,
                              void* d_out, int out_size, void* d_ws, size_t ws_size,
                              hipStream_t stream) {
  (void)in_sizes; (void)n_in; (void)out_size; (void)ws_size;
  const float* memory  = (const float*)d_in[0];
  const float* visual  = (const float*)d_in[1];
  const float* structt = (const float*)d_in[2];
  const float* gstruct = (const float*)d_in[3];
  const float* cam     = (const float*)d_in[4];
  const float* ego_tok = (const float*)d_in[5];
  const float* mot     = (const float*)d_in[6];
  const int*   vismask = (const int*)d_in[7];
  const int*   validm  = (const int*)d_in[8];
  const float* vp_w = (const float*)d_in[9];
  const float* vp_b = (const float*)d_in[10];
  const float* ln_g = (const float*)d_in[11];
  const float* ln_b = (const float*)d_in[12];
  const float* kv_w = (const float*)d_in[13];
  const float* kv_b = (const float*)d_in[14];
  const float* ego_w = (const float*)d_in[15];
  const float* ego_b = (const float*)d_in[16];
  const float* wq = (const float*)d_in[17];
  const float* bq = (const float*)d_in[18];
  const float* wk = (const float*)d_in[19];
  const float* bk = (const float*)d_in[20];
  const float* wv = (const float*)d_in[21];
  const float* bv = (const float*)d_in[22];
  const float* wo = (const float*)d_in[23];
  const float* bo = (const float*)d_in[24];
  const float* un_w = (const float*)d_in[25];
  const float* un_b = (const float*)d_in[26];
  const float* g_wih = (const float*)d_in[27];
  const float* g_whh = (const float*)d_in[28];
  const float* g_bih = (const float*)d_in[29];
  const float* g_bhh = (const float*)d_in[30];
  float* out = (float*)d_out;

  // ---- workspace layout (floats), with safe aliasing to cut footprint ----
  float* ws = (float*)d_ws;
  // Region A (later aliased by gh; gh is written only after last read of A):
  float* vis      = ws;                                   // 4096*256
  float* stats    = vis + (size_t)N_OBJ * D_MEM;          // 4*4096*2
  float* ctx      = stats + (size_t)NHEAD * N_OBJ * 2;    // 4096*256
  float* attn_out = ctx + (size_t)N_OBJ * D_MEM;          // 4096*256
  float* gh       = ws;                                   // aliases region A (3.15M <= 3.18M)
  // Region B (later aliased by gi; gi written only after attention):
  float* q    = attn_out + (size_t)N_OBJ * D_MEM;         // 4096*256
  float* kmat = q + (size_t)N_OBJ * D_MEM;                // 4224*256
  float* vmat = kmat + (size_t)KV_PAD * D_MEM;            // 4224*256
  float* gi   = q;                                        // aliases region B (3.15M <= 3.21M)
  // Independent regions:
  float* kv     = vmat + (size_t)KV_PAD * D_MEM;          // 4224*256
  float* gru_in = kv + (size_t)KV_PAD * D_MEM;            // 4096*704

  dim3 blk(256);

  // 1. visual projector: vis = ReLU(VF @ vp_w + vp_b), then LayerNorm in place
  gemm_bf16_k<<<dim3(N_OBJ / 128, D_MEM / 128), blk, 0, stream>>>(
      visual, D_ROI, vp_w, D_MEM, vp_b, vis, D_MEM, D_ROI, EPI_RELU, nullptr, nullptr);
  layernorm_k<<<dim3(N_OBJ), blk, 0, stream>>>(vis, ln_g, ln_b);

  // 2. KV source rows: global struct proj (rows 0..4095) + ego row + zero pad
  gemm_bf16_k<<<dim3(N_OBJ / 128, D_MEM / 128), blk, 0, stream>>>(
      gstruct, D_STRUCT, kv_w, D_MEM, kv_b, kv, D_MEM, D_STRUCT, EPI_BIAS, nullptr, nullptr);
  ego_pad_k<<<dim3(KV_PAD - N_OBJ), blk, 0, stream>>>(ego_tok, ego_w, ego_b, kv);

  // 3. Q / K / V projections
  gemm_bf16_k<<<dim3(N_OBJ / 128, D_MEM / 128), blk, 0, stream>>>(
      memory, D_MEM, wq, D_MEM, bq, q, D_MEM, D_MEM, EPI_BIAS, nullptr, nullptr);
  gemm_bf16_k<<<dim3(KV_PAD / 128, D_MEM / 128), blk, 0, stream>>>(
      kv, D_MEM, wk, D_MEM, bk, kmat, D_MEM, D_MEM, EPI_BIAS, nullptr, nullptr);
  gemm_bf16_k<<<dim3(KV_PAD / 128, D_MEM / 128), blk, 0, stream>>>(
      kv, D_MEM, wv, D_MEM, bv, vmat, D_MEM, D_MEM, EPI_BIAS, nullptr, nullptr);

  // 4. attention: two-pass lazy softmax (no score matrix materialization)
  attn_stats_k<<<dim3(N_OBJ / 64, NHEAD), blk, 0, stream>>>(q, kmat, stats);
  attn_av_k<<<dim3(N_OBJ / 64, NHEAD), blk, 0, stream>>>(q, kmat, vmat, stats, ctx);

  // 5. output projection
  gemm_bf16_k<<<dim3(N_OBJ / 128, D_MEM / 128), blk, 0, stream>>>(
      ctx, D_MEM, wo, D_MEM, bo, attn_out, D_MEM, D_MEM, EPI_BIAS, nullptr, nullptr);

  // 6. pack struct/cam/mot into gru_input cols 256..703
  pack_k<<<dim3((N_OBJ * 448 + 255) / 256), blk, 0, stream>>>(structt, cam, mot, gru_in);

  // 7. unseen proj + visibility mix -> gru_input cols 0..255
  gemm_bf16_k<<<dim3(N_OBJ / 128, D_MEM / 128), blk, 0, stream>>>(
      attn_out, D_MEM, un_w, D_MEM, un_b, gru_in, GRU_IN, D_MEM, EPI_MIX, vismask, vis);

  // 8. GRU input/hidden GEMMs (gi aliases q/kmat/vmat; gh aliases vis/ctx/attn_out)
  gemm_bf16_k<<<dim3(N_OBJ / 128, GRU_OUT / 128), blk, 0, stream>>>(
      gru_in, GRU_IN, g_wih, GRU_OUT, g_bih, gi, GRU_OUT, GRU_IN, EPI_BIAS, nullptr, nullptr);
  gemm_bf16_k<<<dim3(N_OBJ / 128, GRU_OUT / 128), blk, 0, stream>>>(
      memory, D_MEM, g_whh, GRU_OUT, g_bhh, gh, GRU_OUT, D_MEM, EPI_BIAS, nullptr, nullptr);

  // 9. gate combine + valid mask
  gru_final_k<<<dim3(N_OBJ * D_MEM / 256), blk, 0, stream>>>(gi, gh, memory, validm, out);
}
